// ReinforcedDecoder_69612829933753
// MI455X (gfx1250) — matrix operand order repaired and checked by reading, not compiled, not run on hardware
//
#include <hip/hip_runtime.h>
#include <hip/hip_bf16.h>
#include <math.h>

typedef __bf16 bf16_t;
typedef __bf16 v16bf __attribute__((ext_vector_type(16)));
typedef __bf16 v8bf  __attribute__((ext_vector_type(8)));
typedef float  v8f   __attribute__((ext_vector_type(8)));

#define BSZ   64
#define DIMK  512
#define SRCN  400
#define TDECN 100
#define NEMB  50257
#define CCAT  1536

#define PSTR 520   // proj LDS row stride (halfs): 512 + 8 pad, rows 16B-aligned
#define CSTR 40    // c_cat LDS row stride (halfs)
#define TSTR 40    // tbuf row stride (halfs)

static __device__ inline v8f vzero8() {
  v8f z;
#pragma unroll
  for (int i = 0; i < 8; ++i) z[i] = 0.0f;
  return z;
}
static __device__ inline v16bf cat8(v8bf lo, v8bf hi) {
  return __builtin_shufflevector(lo, hi, 0,1,2,3,4,5,6,7,8,9,10,11,12,13,14,15);
}
static __device__ inline v16bf ldfrag(const bf16_t* p) {
  return cat8(*(const v8bf*)p, *(const v8bf*)(p + 8));
}

// CDNA5 async global->LDS copy, 16B per lane, tracked by ASYNCcnt.
static __device__ inline void async_cp16(void* lds, const void* g) {
  unsigned l = (unsigned)(uintptr_t)lds;   // low 32 bits of generic LDS addr = LDS offset
  asm volatile("global_load_async_to_lds_b128 %0, %1, off" :: "v"(l), "v"(g) : "memory");
}
static __device__ inline void wait_async0() {
  asm volatile("s_wait_asynccnt 0x0" ::: "memory");
}

// ---------------------------------------------------------------------------
// K0: W_proj [512][1536] fp32 -> projT [1536][512] bf16 (transposed)
// ---------------------------------------------------------------------------
__global__ void k_proj_cvt(const float* __restrict__ Wp, bf16_t* __restrict__ projT) {
  int idx = blockIdx.x * 256 + threadIdx.x;       // idx = j*512 + k
  if (idx >= CCAT * DIMK) return;
  int j = idx >> 9, k = idx & 511;
  projT[idx] = (bf16_t)Wp[k * CCAT + j];
}

// ---------------------------------------------------------------------------
// K1: q = h_t @ W_attn  (enc + dec). block = (mat*64 + b)
// ---------------------------------------------------------------------------
__global__ void k_q(const float* __restrict__ ht, const float* __restrict__ We,
                    const float* __restrict__ Wd, float* __restrict__ qws) {
  int b = blockIdx.x & 63, mat = blockIdx.x >> 6;
  const float* W = mat ? Wd : We;
  __shared__ float hl[DIMK];
  for (int i = threadIdx.x; i < DIMK; i += 256) hl[i] = ht[b * DIMK + i];
  __syncthreads();
  for (int d = threadIdx.x; d < DIMK; d += 256) {
    float acc = 0.0f;
    for (int k = 0; k < DIMK; ++k) acc = fmaf(hl[k], W[k * DIMK + d], acc);
    qws[(mat * BSZ + b) * DIMK + d] = acc;
  }
}

// ---------------------------------------------------------------------------
// K2: fused intra-attention: E = q . hv ; A = E / sum(E) ; C = hv @ A
// hv[b] = flat + b*DIMK*nl (raw .view reinterpretation), row stride = nl
// ---------------------------------------------------------------------------
__global__ void k_attn(const float* __restrict__ henc, const float* __restrict__ hdec,
                       const float* __restrict__ qws, float* __restrict__ ae,
                       float* __restrict__ ce, float* __restrict__ cd,
                       bf16_t* __restrict__ ccatT) {
  int b = blockIdx.x & 63, mat = blockIdx.x >> 6;
  int nl = mat ? TDECN : SRCN;
  const float* hv = (mat ? hdec : henc) + (size_t)b * DIMK * nl;

  __shared__ float ql[DIMK];
  __shared__ float Al[SRCN];
  __shared__ float red[256];

  for (int i = threadIdx.x; i < DIMK; i += 256)
    ql[i] = qws[(mat * BSZ + b) * DIMK + i];
  __syncthreads();

  float part = 0.0f;
  for (int n = threadIdx.x; n < nl; n += 256) {
    float e = 0.0f;
    for (int d = 0; d < DIMK; ++d) e = fmaf(ql[d], hv[(size_t)d * nl + n], e);
    Al[n] = e;
    part += e;
  }
  red[threadIdx.x] = part;
  __syncthreads();
  for (int s = 128; s > 0; s >>= 1) {
    if (threadIdx.x < s) red[threadIdx.x] += red[threadIdx.x + s];
    __syncthreads();
  }
  float inv = 1.0f / red[0];
  for (int n = threadIdx.x; n < nl; n += 256) {
    float a = Al[n] * inv;
    Al[n] = a;
    if (mat == 0) ae[b * SRCN + n] = a;
  }
  __syncthreads();

  for (int d = threadIdx.x; d < DIMK; d += 256) {
    float c = 0.0f;
    for (int n = 0; n < nl; ++n) c = fmaf(hv[(size_t)d * nl + n], Al[n], c);
    if (mat == 0) { ce[b * DIMK + d] = c; ccatT[b * CCAT + DIMK + d]     = (bf16_t)c; }
    else          { cd[b * DIMK + d] = c; ccatT[b * CCAT + 2 * DIMK + d] = (bf16_t)c; }
  }
}

// ---------------------------------------------------------------------------
// K3: c_catT[b][0:512] = bf16(h_t[b]);  p_switch[b] = sigmoid(W_u . c_cat + b_u)
// ---------------------------------------------------------------------------
__global__ void k_pswitch(const float* __restrict__ ht, const float* __restrict__ ce,
                          const float* __restrict__ cd, const float* __restrict__ Wu,
                          const float* __restrict__ bu, bf16_t* __restrict__ ccatT,
                          float* __restrict__ ps) {
  int b = blockIdx.x;
  __shared__ float red[256];
  float part = 0.0f;
  for (int j = threadIdx.x; j < CCAT; j += 256) {
    float v;
    if (j < DIMK)          { v = ht[b * DIMK + j]; ccatT[b * CCAT + j] = (bf16_t)v; }
    else if (j < 2 * DIMK) { v = ce[b * DIMK + j - DIMK]; }
    else                   { v = cd[b * DIMK + j - 2 * DIMK]; }
    part = fmaf(Wu[j], v, part);
  }
  red[threadIdx.x] = part;
  __syncthreads();
  for (int s = 128; s > 0; s >>= 1) {
    if (threadIdx.x < s) red[threadIdx.x] += red[threadIdx.x + s];
    __syncthreads();
  }
  if (threadIdx.x == 0) ps[b] = 1.0f / (1.0f + __expf(-(red[0] + bu[0])));
}

// ---------------------------------------------------------------------------
// K4: fused logits = tanh(W_emb @ W_proj) @ c_cat + b_out  via WMMA bf16.
// 128 threads = 4 waves = 1 wave/SIMD32 (~450 VGPRs/wave is safe).
// Each wave owns 32 vocab rows (2 M-tiles) -> every LDS B-fragment feeds 2
// WMMAs. W_emb A-fragments live in registers for all 48 chunks.
// Double-buffered async global->LDS staging + double-buffered B fragments.
// ---------------------------------------------------------------------------
__global__ __launch_bounds__(128)
void k_gemm(const float* __restrict__ Wemb, const bf16_t* __restrict__ projT,
            const bf16_t* __restrict__ ccatT, const float* __restrict__ bout,
            float* __restrict__ logits) {
  __shared__ __align__(16) bf16_t proj_lds[2][32 * PSTR];  // 2 x 33.3KB
  __shared__ __align__(16) bf16_t ccat_lds[2][64 * CSTR];  // 2 x 5.1KB
  __shared__ __align__(16) bf16_t tbuf[4][32 * TSTR];      // per-wave, 10.2KB

  const int tid  = threadIdx.x;
  const int wave = tid >> 5;
  const int lane = tid & 31;
  const int r    = lane & 15;
  const int hi   = lane >> 4;
  const int m0   = blockIdx.x * 128 + wave * 32;

  // Preload W_emb A fragments (2 M-tiles x 16 K-steps) into registers.
  // A 16x32 bf16 layout: lane = M row (r); halfs 0-7: K in [kb+8*hi, +8),
  //                                        halfs 8-15: K in [kb+16+8*hi, +8)
  v16bf aReg[2][16];
#pragma unroll
  for (int mt = 0; mt < 2; ++mt) {
    int rowg = m0 + mt * 16 + r;
    if (rowg > NEMB - 1) rowg = NEMB - 1;
    const float* wrow = Wemb + (size_t)rowg * DIMK;
#pragma unroll
    for (int ks = 0; ks < 16; ++ks) {
      int kb = ks * 32;
      const float4* p0 = (const float4*)(wrow + kb + 8 * hi);
      const float4* p1 = (const float4*)(wrow + kb + 16 + 8 * hi);
      float4 x0 = p0[0], x1 = p0[1], y0 = p1[0], y1 = p1[1];
      v16bf a;
      a[0]=(bf16_t)x0.x; a[1]=(bf16_t)x0.y; a[2]=(bf16_t)x0.z; a[3]=(bf16_t)x0.w;
      a[4]=(bf16_t)x1.x; a[5]=(bf16_t)x1.y; a[6]=(bf16_t)x1.z; a[7]=(bf16_t)x1.w;
      a[8]=(bf16_t)y0.x; a[9]=(bf16_t)y0.y; a[10]=(bf16_t)y0.z; a[11]=(bf16_t)y0.w;
      a[12]=(bf16_t)y1.x; a[13]=(bf16_t)y1.y; a[14]=(bf16_t)y1.z; a[15]=(bf16_t)y1.w;
      aReg[mt][ks] = a;
    }
  }

  const char* projB = (const char*)projT;
  const char* ccatB = (const char*)ccatT;

  // Async-stage one j-chunk (32 columns of W_proj + 32 rows of c_catT) into buf.
  auto stage_async = [&](int jc, int buf) {
#pragma unroll
    for (int i = 0; i < 16; ++i) {                   // proj: 2048 x 16B chunks
      int cid  = i * 128 + tid;
      int n    = cid >> 6;                           // chunk column 0..31
      int part = cid & 63;                           // 16B piece within 1KB row
      async_cp16(&proj_lds[buf][n * PSTR + part * 8],
                 projB + ((size_t)(jc * 32 + n) * DIMK + part * 8) * 2);
    }
#pragma unroll
    for (int i = 0; i < 2; ++i) {                    // ccat: 256 x 16B chunks
      int cid = i * 128 + tid;
      int b = cid >> 2, part = cid & 3;
      async_cp16(&ccat_lds[buf][b * CSTR + part * 8],
                 ccatB + ((size_t)b * CCAT + jc * 32 + part * 8) * 2);
    }
  };

  stage_async(0, 0);

  v8f L[2][4];
#pragma unroll
  for (int mt = 0; mt < 2; ++mt)
#pragma unroll
    for (int bt = 0; bt < 4; ++bt) L[mt][bt] = vzero8();

  for (int jc = 0; jc < 48; ++jc) {
    const int cur = jc & 1;
    wait_async0();        // our async writes into buf[cur] have landed in LDS
    __syncthreads();      // everyone's writes visible; prev reads of buf[1-cur] done
    if (jc + 1 < 48) stage_async(jc + 1, cur ^ 1);

    const bf16_t* P = proj_lds[cur];
    const bf16_t* C = ccat_lds[cur];

    // Prefetch GEMM2 B-fragments now; they're only consumed after the K-loop,
    // so their LDS latency is fully hidden behind 64 WMMAs.
    v16bf Bc[4];
#pragma unroll
    for (int bt = 0; bt < 4; ++bt)
      Bc[bt] = cat8(*(const v8bf*)(C + (bt * 16 + r) * CSTR + 8 * hi), *(const v8bf*)(C + (bt * 16 + r) * CSTR + 16 + 8 * hi));

    // GEMM1: T = W_emb_tile @ W_proj[:, chunk]  (M32 x N32, K=512) with
    // double-buffered B fragments (load ks+1 before issuing ks's 4 WMMAs).
    const bf16_t* brow0 = P + r * PSTR + 8 * hi;         // B column r
    const bf16_t* brow1 = P + (16 + r) * PSTR + 8 * hi;  // B column 16+r
    v8f t00 = vzero8(), t01 = vzero8(), t10 = vzero8(), t11 = vzero8();
    v16bf B0 = cat8(*(const v8bf*)brow0, *(const v8bf*)(brow0 + 16)), B1 = cat8(*(const v8bf*)brow1, *(const v8bf*)(brow1 + 16));
#pragma unroll
    for (int ks = 0; ks < 16; ++ks) {
      v16bf B0n, B1n;
      if (ks < 15) {
        int kb = (ks + 1) * 32;
        B0n = cat8(*(const v8bf*)(brow0 + kb), *(const v8bf*)(brow0 + kb + 16));
        B1n = cat8(*(const v8bf*)(brow1 + kb), *(const v8bf*)(brow1 + kb + 16));
      }
      t00 = __builtin_amdgcn_wmma_f32_16x16x32_bf16(false, aReg[0][ks], false, B0, (short)0, t00, false, false);
      t01 = __builtin_amdgcn_wmma_f32_16x16x32_bf16(false, aReg[0][ks], false, B1, (short)0, t01, false, false);
      t10 = __builtin_amdgcn_wmma_f32_16x16x32_bf16(false, aReg[1][ks], false, B0, (short)0, t10, false, false);
      t11 = __builtin_amdgcn_wmma_f32_16x16x32_bf16(false, aReg[1][ks], false, B1, (short)0, t11, false, false);
      if (ks < 15) { B0 = B0n; B1 = B1n; }
    }

    // tanh + C-frag -> A-frag relayout through per-wave LDS buffer tbuf[m][j].
    // Wave-local LDS ops are processed in order (DScnt), so no workgroup
    // barrier is needed -- only a compiler ordering fence.
    bf16_t* tb = tbuf[wave];
#pragma unroll
    for (int qq = 0; qq < 8; ++qq) {
      tb[(qq + 8 * hi) * TSTR + r]           = (bf16_t)tanhf(t00[qq]);
      tb[(qq + 8 * hi) * TSTR + r + 16]      = (bf16_t)tanhf(t01[qq]);
      tb[(16 + qq + 8 * hi) * TSTR + r]      = (bf16_t)tanhf(t10[qq]);
      tb[(16 + qq + 8 * hi) * TSTR + r + 16] = (bf16_t)tanhf(t11[qq]);
    }
    asm volatile("" ::: "memory");

    const bf16_t* tp0 = tb + r * TSTR;
    const bf16_t* tp1 = tb + (16 + r) * TSTR;
    v16bf A20 = cat8(*(const v8bf*)(tp0 + 8 * hi), *(const v8bf*)(tp0 + 16 + 8 * hi));
    v16bf A21 = cat8(*(const v8bf*)(tp1 + 8 * hi), *(const v8bf*)(tp1 + 16 + 8 * hi));

    // GEMM2: logits += T_chunk @ c_cat[chunk, 0:64]  (4 batch N-tiles, B reused
    // by both M-tiles, fragments prefetched above)
#pragma unroll
    for (int bt = 0; bt < 4; ++bt) {
      L[0][bt] = __builtin_amdgcn_wmma_f32_16x16x32_bf16(false, A20, false, Bc[bt], (short)0, L[0][bt], false, false);
      L[1][bt] = __builtin_amdgcn_wmma_f32_16x16x32_bf16(false, A21, false, Bc[bt], (short)0, L[1][bt], false, false);
    }
  }

  // Store logits tile: C/D layout -> logits[m][b], add b_out[m]
#pragma unroll
  for (int mt = 0; mt < 2; ++mt) {
#pragma unroll
    for (int qq = 0; qq < 8; ++qq) {
      int m = m0 + mt * 16 + qq + 8 * hi;
      if (m < NEMB) {
        float bo = bout[m];
        logits[(size_t)m * 64 + 0  + r] = L[mt][0][qq] + bo;
        logits[(size_t)m * 64 + 16 + r] = L[mt][1][qq] + bo;
        logits[(size_t)m * 64 + 32 + r] = L[mt][2][qq] + bo;
        logits[(size_t)m * 64 + 48 + r] = L[mt][3][qq] + bo;
      }
    }
  }
}

// ---------------------------------------------------------------------------
// K5: per-batch-column online softmax stats over the vocab
// ---------------------------------------------------------------------------
__global__ void k_stats(const float* __restrict__ logits, float* __restrict__ Mb,
                        float* __restrict__ Sb) {
  int b = blockIdx.x;
  __shared__ float rm[256], rs[256];
  float m = -INFINITY, s = 0.0f;
  for (int v = threadIdx.x; v < NEMB; v += 256) {
    float x = logits[(size_t)v * 64 + b];
    if (x > m) { s *= __expf(m - x); m = x; }
    s += __expf(x - m);
  }
  rm[threadIdx.x] = m; rs[threadIdx.x] = s;
  __syncthreads();
  for (int st = 128; st > 0; st >>= 1) {
    if (threadIdx.x < st) {
      float m1 = rm[threadIdx.x], s1 = rs[threadIdx.x];
      float m2 = rm[threadIdx.x + st], s2 = rs[threadIdx.x + st];
      float M = fmaxf(m1, m2);
      rm[threadIdx.x] = M;
      rs[threadIdx.x] = s1 * __expf(m1 - M) + s2 * __expf(m2 - M);
    }
    __syncthreads();
  }
  if (threadIdx.x == 0) { Mb[b] = rm[0]; Sb[b] = rs[0]; }
}

// ---------------------------------------------------------------------------
// K6: p_gen[b][v] = (1 - p_switch[b]) * softmax(logits)[v][b]
// ---------------------------------------------------------------------------
__global__ void k_pgen(const float* __restrict__ logits, const float* __restrict__ Mb,
                       const float* __restrict__ Sb, const float* __restrict__ ps,
                       float* __restrict__ out) {
  int b = blockIdx.y;
  int v = blockIdx.x * 256 + threadIdx.x;
  if (v >= NEMB) return;
  float p = (1.0f - ps[b]) * __expf(logits[(size_t)v * 64 + b] - Mb[b]) / Sb[b];
  out[(size_t)b * NEMB + v] = p;
}

// ---------------------------------------------------------------------------
// K7: p_copy[b][s] = p_switch[b] * ae[b][s]
// ---------------------------------------------------------------------------
__global__ void k_pcopy(const float* __restrict__ ae, const float* __restrict__ ps,
                        float* __restrict__ out2) {
  int idx = blockIdx.x * 256 + threadIdx.x;
  if (idx >= BSZ * SRCN) return;
  int b = idx / SRCN;
  out2[idx] = ps[b] * ae[idx];
}

// ---------------------------------------------------------------------------
extern "C" void kernel_launch(void* const* d_in, const int* in_sizes, int n_in,
                              void* d_out, int out_size, void* d_ws, size_t ws_size,
                              hipStream_t stream) {
  const float* h_t   = (const float*)d_in[0];
  const float* h_enc = (const float*)d_in[1];
  const float* h_dec = (const float*)d_in[2];
  // d_in[3] = x (token ids) — unused by the reference math
  const float* Wae   = (const float*)d_in[4];
  const float* Wad   = (const float*)d_in[5];
  const float* Wemb  = (const float*)d_in[6];
  const float* Wproj = (const float*)d_in[7];
  const float* Wu    = (const float*)d_in[8];
  const float* bu    = (const float*)d_in[9];
  const float* bout  = (const float*)d_in[10];

  char* ws = (char*)d_ws;
  bf16_t* projT  = (bf16_t*)(ws + 0);          // 1536*512*2  = 1,572,864
  bf16_t* ccatT  = (bf16_t*)(ws + 1572864);    // 64*1536*2   =   196,608
  float*  qws    = (float*)(ws + 1769472);     // 2*64*512*4  =   262,144
  float*  ae     = (float*)(ws + 2031616);     // 64*400*4    =   102,400
  float*  ce     = (float*)(ws + 2134016);     // 64*512*4    =   131,072
  float*  cd     = (float*)(ws + 2265088);     // 64*512*4    =   131,072
  float*  ps     = (float*)(ws + 2396160);     // 64*4 (padded)
  float*  Mb     = (float*)(ws + 2396416);
  float*  Sb     = (float*)(ws + 2396672);
  float*  logits = (float*)(ws + 2396928);     // 50257*64*4 = 12,865,792

  float* out_pgen  = (float*)d_out;
  float* out_pcopy = out_pgen + (size_t)BSZ * NEMB;

  k_proj_cvt<<<(CCAT * DIMK) / 256, 256, 0, stream>>>(Wproj, projT);
  k_q       <<<128, 256, 0, stream>>>(h_t, Wae, Wad, qws);
  k_attn    <<<128, 256, 0, stream>>>(h_enc, h_dec, qws, ae, ce, cd, ccatT);
  k_pswitch <<<64, 256, 0, stream>>>(h_t, ce, cd, Wu, bu, ccatT, ps);
  k_gemm    <<<(NEMB + 127) / 128, 128, 0, stream>>>(Wemb, projT, ccatT, bout, logits);
  k_stats   <<<64, 256, 0, stream>>>(logits, Mb, Sb);
  k_pgen    <<<dim3((NEMB + 255) / 256, BSZ), 256, 0, stream>>>(logits, Mb, Sb, ps, out_pgen);
  k_pcopy   <<<(BSZ * SRCN + 255) / 256, 256, 0, stream>>>(ae, ps, out_pcopy);
}
